// ASTEncoder_81750407512811
// MI455X (gfx1250) — matrix operand order repaired and checked by reading, not compiled, not run on hardware
//
#include <hip/hip_runtime.h>
#include <hip/hip_bf16.h>
#include <math.h>

typedef __attribute__((ext_vector_type(16))) _Float16 v16h;
typedef __attribute__((ext_vector_type(8)))  _Float16 v8h;
typedef __attribute__((ext_vector_type(4)))  _Float16 v4h;
typedef __attribute__((ext_vector_type(8)))  float    v8f;

#define BM 128
#define BN 128
#define BK 32

__device__ __forceinline__ float gelu_tanh(float x) {
    float x3 = x * x * x;
    return 0.5f * x * (1.0f + tanhf(0.7978845608028654f * (x + 0.044715f * x3)));
}

// ---------------------------------------------------------------------------
// GEMM: C[M,N] = A(f16)[M,K] * B(f32->f16)[K,N] + bias, with epilogue MODE
//   MODE 0: out32[m,n]  = acc + bias[n]            (QKV projections)
//   MODE 1: out32[m,n] += acc + bias[n]            (residual add, in place)
//   MODE 2: out16[m,n]  = f16(gelu(acc + bias[n])) (FF1)
// Block: 256 threads = 8 waves. Block tile 128x128, K-step 32.
// Wave grid 4(m) x 2(n); per-wave tile 32x64 = 2x4 WMMA 16x16 fragments.
// Register double-buffer: tile k+1 is fetched from global into VGPRs while
// the WMMA chain consumes tile k from LDS (global latency hidden).
// ---------------------------------------------------------------------------
template<int MODE>
__global__ __launch_bounds__(256)
void gemm_wmma_kernel(const _Float16* __restrict__ A,
                      const float*    __restrict__ Bw,
                      const float*    __restrict__ bias,
                      float* __restrict__ outF32,
                      _Float16* __restrict__ outF16,
                      int M, int N, int K)
{
    __shared__ __align__(16) _Float16 As[BM * BK];   // [m][k], stride BK
    __shared__ __align__(32) _Float16 Bs[BN * BK];   // transposed: [n][k], stride BK

    const int tid  = threadIdx.x;
    const int lane = tid & 31;
    const int wave = tid >> 5;
    const int wm   = wave & 3;          // 0..3 -> 32-row slab
    const int wn   = wave >> 2;         // 0..1 -> 64-col slab
    const int rowBase = blockIdx.y * BM;
    const int colBase = blockIdx.x * BN;

    // Fragment lane mapping (CDNA5 ISA 7.12.2, 16-bit A 16x32 / B 32x16):
    const int mrow  = lane & 15;
    const int khalf = (lane >> 4) * 8;   // A: elems 0-7 -> K=khalf.., 8-15 -> K=16+khalf..
    const int koff  = (lane >> 4) * 16;  // unused; B reads use khalf, same k order as A
    const int ncol  = lane & 15;

    // A staging: 4 chunks of 4 halves (8B) per thread
    const int ar[4] = { (tid + 0)   >> 3, (tid + 256) >> 3,
                        (tid + 512) >> 3, (tid + 768) >> 3 };
    const int ac    = (tid & 7) * 4;
    // B staging: 2 pair-chunks; each = 2 adjacent k-rows x 4 n-cols
    const int bp[2] = { tid >> 5, (tid + 256) >> 5 };       // k-pair 0..15
    const int bc[2] = { (tid & 31) * 4, (tid & 31) * 4 };   // n-col

    v4h    areg[4];
    float4 breg[2][2];

    auto load_tile = [&](int k0) {
        #pragma unroll
        for (int qq = 0; qq < 4; ++qq)
            areg[qq] = *(const v4h*)(A + (size_t)(rowBase + ar[qq]) * K + k0 + ac);
        #pragma unroll
        for (int qq = 0; qq < 2; ++qq) {
            const float* bsrc = Bw + (size_t)(k0 + 2 * bp[qq]) * N + colBase + bc[qq];
            breg[qq][0] = *(const float4*)(bsrc);
            breg[qq][1] = *(const float4*)(bsrc + N);
        }
    };
    auto store_tile = [&]() {
        #pragma unroll
        for (int qq = 0; qq < 4; ++qq)
            *(v4h*)&As[ar[qq] * BK + ac] = areg[qq];
        #pragma unroll
        for (int qq = 0; qq < 2; ++qq) {
            const float f0[4] = { breg[qq][0].x, breg[qq][0].y, breg[qq][0].z, breg[qq][0].w };
            const float f1[4] = { breg[qq][1].x, breg[qq][1].y, breg[qq][1].z, breg[qq][1].w };
            #pragma unroll
            for (int j = 0; j < 4; ++j) {
                union { _Float16 h[2]; unsigned u; } pk;
                pk.h[0] = (_Float16)f0[j];           // k = 2p
                pk.h[1] = (_Float16)f1[j];           // k = 2p+1
                *(unsigned*)&Bs[(bc[qq] + j) * BK + 2 * bp[qq]] = pk.u;
            }
        }
    };

    v8f acc[2][4];
    #pragma unroll
    for (int i = 0; i < 2; ++i)
        #pragma unroll
        for (int j = 0; j < 4; ++j)
            acc[i][j] = (v8f)0.0f;

    load_tile(0);                                   // prefetch tile 0

    for (int k0 = 0; k0 < K; k0 += BK) {
        store_tile();
        __syncthreads();
        if (k0 + BK < K)
            load_tile(k0 + BK);                     // prefetch next tile into regs

        union { v16h v; v8h h[2]; } af[2];
        #pragma unroll
        for (int mt = 0; mt < 2; ++mt) {
            int row = wm * 32 + mt * 16 + mrow;
            af[mt].h[0] = *(const v8h*)&As[row * BK + khalf];
            af[mt].h[1] = *(const v8h*)&As[row * BK + 16 + khalf];
        }
        #pragma unroll
        for (int nt = 0; nt < 4; ++nt) {
            int col = wn * 64 + nt * 16 + ncol;
            union { v16h v; v8h h[2]; } bu;
            bu.h[0] = *(const v8h*)&Bs[col * BK + khalf];
            bu.h[1] = *(const v8h*)&Bs[col * BK + 16 + khalf];
            v16h bf = bu.v;
            #pragma unroll
            for (int mt = 0; mt < 2; ++mt) {
                acc[mt][nt] = __builtin_amdgcn_wmma_f32_16x16x32_f16(
                    false, af[mt].v, false, bf, (short)0, acc[mt][nt], false, false);
            }
        }
        __syncthreads();
    }

    // --- epilogue: C layout VGPR j -> row j + (lane>=16)*8, col lane&15 ---
    #pragma unroll
    for (int mt = 0; mt < 2; ++mt) {
        #pragma unroll
        for (int nt = 0; nt < 4; ++nt) {
            int gm0 = rowBase + wm * 32 + mt * 16 + (lane >> 4) * 8;
            int gn  = colBase + wn * 64 + nt * 16 + (lane & 15);
            float bb = bias[gn];
            #pragma unroll
            for (int j = 0; j < 8; ++j) {
                float val = acc[mt][nt][j] + bb;
                size_t gidx = (size_t)(gm0 + j) * N + gn;
                if (MODE == 0)      outF32[gidx] = val;
                else if (MODE == 1) outF32[gidx] = outF32[gidx] + val;
                else                outF16[gidx] = (_Float16)gelu_tanh(val);
            }
        }
    }
}

// ---------------------------------------------------------------------------
// LayerNorm over D=1024, one block per row.  OUT16=1 -> f16 activations.
// ---------------------------------------------------------------------------
template<int OUT16>
__global__ __launch_bounds__(256)
void ln_kernel(const float* __restrict__ x,
               const float* __restrict__ g,
               const float* __restrict__ b,
               _Float16* __restrict__ o16,
               float* __restrict__ o32)
{
    const int D = 1024;
    const int row = blockIdx.x;
    const int tid = threadIdx.x;
    __shared__ float s_sum[256];
    __shared__ float s_sq[256];
    const float* xr = x + (size_t)row * D;

    float v[4];
    float sum = 0.0f, sq = 0.0f;
    #pragma unroll
    for (int i = 0; i < 4; ++i) {
        v[i] = xr[tid + i * 256];
        sum += v[i];
        sq  += v[i] * v[i];
    }
    s_sum[tid] = sum; s_sq[tid] = sq;
    __syncthreads();
    #pragma unroll
    for (int off = 128; off > 0; off >>= 1) {
        if (tid < off) { s_sum[tid] += s_sum[tid + off]; s_sq[tid] += s_sq[tid + off]; }
        __syncthreads();
    }
    float mu   = s_sum[0] * (1.0f / D);
    float var  = s_sq[0] * (1.0f / D) - mu * mu;
    float rstd = rsqrtf(var + 1e-5f);

    #pragma unroll
    for (int i = 0; i < 4; ++i) {
        int c = tid + i * 256;
        float y = (v[i] - mu) * rstd * g[c] + b[c];
        if (OUT16) o16[(size_t)row * D + c] = (_Float16)y;
        else       o32[(size_t)row * D + c] = y;
    }
}

// ---------------------------------------------------------------------------
// Tree-relative attention: one wave per (b,h,l); R=16 gathered neighbors,
// dk=64 split 2 dims/lane; wave-shuffle dot reduction; softmax in registers.
// Writes ctx directly as f16 (input to the out-projection WMMA GEMM).
// ---------------------------------------------------------------------------
__global__ __launch_bounds__(256)
void attn_kernel(const float* __restrict__ q,
                 const float* __restrict__ k,
                 const float* __restrict__ v,
                 const int*   __restrict__ pos,
                 const float* __restrict__ rel_q,
                 const float* __restrict__ rel_k,
                 const float* __restrict__ rel_v,
                 _Float16* __restrict__ ctx16)
{
    const int L = 1024, H = 16, R = 16, dk = 64, D = 1024;
    const int wave = threadIdx.x >> 5;
    const int lane = threadIdx.x & 31;
    const int item = blockIdx.x * 8 + wave;     // (b*H + h)*L + l
    const int l  = item & (L - 1);
    const int bh = item >> 10;
    const int h  = bh & (H - 1);
    const int b  = bh >> 4;
    const int d0 = lane, d1 = lane + 32;

    const size_t qbase = ((size_t)(b * L + l)) * D + h * dk;
    const float q0 = q[qbase + d0];
    const float q1 = q[qbase + d1];

    float scores[16];
    int   idxs[16];
    #pragma unroll
    for (int r = 0; r < R; ++r) {
        int idx = pos[((size_t)bh * R + r) * L + l];
        idxs[r] = idx;
        size_t kb = ((size_t)(b * L + idx)) * D + h * dk;
        size_t rb = ((size_t)(h * R + r)) * dk;
        // (q + rel_q)·k_g + q·rel_k  ==  q·k_g + rel_q·k_g + q·rel_k
        float p = (q0 + rel_q[rb + d0]) * k[kb + d0]
                + (q1 + rel_q[rb + d1]) * k[kb + d1]
                + q0 * rel_k[rb + d0]
                + q1 * rel_k[rb + d1];
        #pragma unroll
        for (int off = 16; off > 0; off >>= 1)
            p += __shfl_xor(p, off, 32);
        float s = p * 0.125f;                    // 1/sqrt(64)
        scores[r] = (idx == l) ? -1e9f : s;      // padding sentinel mask
    }

    float mx = scores[0];
    #pragma unroll
    for (int r = 1; r < R; ++r) mx = fmaxf(mx, scores[r]);
    float w[16];
    float denom = 0.0f;
    #pragma unroll
    for (int r = 0; r < R; ++r) { w[r] = __expf(scores[r] - mx); denom += w[r]; }
    float inv = 1.0f / denom;

    float c0 = 0.0f, c1 = 0.0f;
    #pragma unroll
    for (int r = 0; r < R; ++r) {
        size_t vb = ((size_t)(b * L + idxs[r])) * D + h * dk;
        size_t rb = ((size_t)(h * R + r)) * dk;
        float wr = w[r] * inv;
        c0 += wr * (v[vb + d0] + rel_v[rb + d0]);
        c1 += wr * (v[vb + d1] + rel_v[rb + d1]);
    }
    ctx16[qbase + d0] = (_Float16)c0;
    ctx16[qbase + d1] = (_Float16)c1;
}

// ---------------------------------------------------------------------------
extern "C" void kernel_launch(void* const* d_in, const int* in_sizes, int n_in,
                              void* d_out, int out_size, void* d_ws, size_t ws_size,
                              hipStream_t stream)
{
    const int B = 2, L = 1024, D = 1024, H = 16, F = 4096, NL = 4;
    const int M = B * L;                       // 2048 rows of the token stream

    const float* emb    = (const float*)d_in[0];
    const int*   pos    = (const int*)  d_in[1];
    const float* rel_q  = (const float*)d_in[2];
    const float* rel_k  = (const float*)d_in[3];
    const float* rel_v  = (const float*)d_in[4];
    const float* attn_w = (const float*)d_in[5];
    const float* attn_b = (const float*)d_in[6];
    const float* ff_w1  = (const float*)d_in[7];
    const float* ff_b1  = (const float*)d_in[8];
    const float* ff_w2  = (const float*)d_in[9];
    const float* ff_b2  = (const float*)d_in[10];
    const float* ln_g   = (const float*)d_in[11];
    const float* ln_b   = (const float*)d_in[12];
    const float* fin_g  = (const float*)d_in[13];
    const float* fin_b  = (const float*)d_in[14];

    // workspace layout (~56 MB)
    char* ws = (char*)d_ws;
    size_t off = 0;
    auto alloc = [&](size_t bytes) -> void* {
        void* p = ws + off;
        off = (off + bytes + 255) & ~(size_t)255;
        return p;
    };
    float*    x     = (float*)   alloc((size_t)M * D * 4);
    _Float16* h16   = (_Float16*)alloc((size_t)M * D * 2);
    float*    qbuf  = (float*)   alloc((size_t)M * D * 4);
    float*    kbuf  = (float*)   alloc((size_t)M * D * 4);
    float*    vbuf  = (float*)   alloc((size_t)M * D * 4);
    _Float16* ctx16 = (_Float16*)alloc((size_t)M * D * 2);
    _Float16* ff16  = (_Float16*)alloc((size_t)M * F * 2);
    (void)ws_size; (void)in_sizes; (void)n_in; (void)out_size;

    // residual stream x <- emb
    hipMemcpyAsync(x, emb, (size_t)M * D * sizeof(float),
                   hipMemcpyDeviceToDevice, stream);

    dim3 blk(256);
    dim3 gD(D / BN, M / BM);   // (8,16)
    dim3 gF(F / BN, M / BM);   // (32,16)

    for (int i = 0; i < NL; ++i) {
        // --- attention sublayer ---
        ln_kernel<1><<<M, blk, 0, stream>>>(
            x, ln_g + (size_t)(i * 2 + 0) * D, ln_b + (size_t)(i * 2 + 0) * D,
            h16, nullptr);
        gemm_wmma_kernel<0><<<gD, blk, 0, stream>>>(
            h16, attn_w + (size_t)(i * 4 + 0) * D * D,
            attn_b + (size_t)(i * 4 + 0) * D, qbuf, nullptr, M, D, D);
        gemm_wmma_kernel<0><<<gD, blk, 0, stream>>>(
            h16, attn_w + (size_t)(i * 4 + 1) * D * D,
            attn_b + (size_t)(i * 4 + 1) * D, kbuf, nullptr, M, D, D);
        gemm_wmma_kernel<0><<<gD, blk, 0, stream>>>(
            h16, attn_w + (size_t)(i * 4 + 2) * D * D,
            attn_b + (size_t)(i * 4 + 2) * D, vbuf, nullptr, M, D, D);
        attn_kernel<<<(B * H * L) / 8, blk, 0, stream>>>(
            qbuf, kbuf, vbuf, pos, rel_q, rel_k, rel_v, ctx16);
        gemm_wmma_kernel<1><<<gD, blk, 0, stream>>>(
            ctx16, attn_w + (size_t)(i * 4 + 3) * D * D,
            attn_b + (size_t)(i * 4 + 3) * D, x, nullptr, M, D, D);

        // --- feed-forward sublayer ---
        ln_kernel<1><<<M, blk, 0, stream>>>(
            x, ln_g + (size_t)(i * 2 + 1) * D, ln_b + (size_t)(i * 2 + 1) * D,
            h16, nullptr);
        gemm_wmma_kernel<2><<<gF, blk, 0, stream>>>(
            h16, ff_w1 + (size_t)i * D * F, ff_b1 + (size_t)i * F,
            nullptr, ff16, M, F, D);
        gemm_wmma_kernel<1><<<gD, blk, 0, stream>>>(
            ff16, ff_w2 + (size_t)i * F * D, ff_b2 + (size_t)i * D,
            x, nullptr, M, D, F);
    }

    ln_kernel<0><<<M, blk, 0, stream>>>(x, fin_g, fin_b, nullptr, (float*)d_out);
}
